// WindowAttention_81724637708589
// MI455X (gfx1250) — matrix operand order repaired and checked by reading, not compiled, not run on hardware
//
#include <hip/hip_runtime.h>
#include <hip/hip_bf16.h>

// ---------------------------------------------------------------------------
// Window attention, CDNA5 (gfx1250), wave32, v_wmma_f32_16x16x32_bf16 path.
// ---------------------------------------------------------------------------

typedef __attribute__((ext_vector_type(16))) __bf16 v16bf;
typedef __attribute__((ext_vector_type(8)))  float  v8f;

union FragBF {
  v16bf  v;
  uint4  q[2];
  __bf16 h[16];
};

static constexpr int BB    = 4;
static constexpr int NWIN  = 512;
static constexpr int L     = 64;     // 8x8 window
static constexpr int CIN   = 256;
static constexpr int COUT  = 256;
static constexpr int H     = 8;
static constexpr int HD    = 32;     // head dim
static constexpr int NPOS  = 225;    // (2*8-1)^2
static constexpr int MROWS = BB * NWIN * L;     // 131072
static constexpr int NQKV  = 3 * COUT;          // 768

__device__ __forceinline__ v8f zero8() {
  v8f z;
#pragma unroll
  for (int i = 0; i < 8; ++i) z[i] = 0.0f;
  return z;
}

__device__ __forceinline__ v8f wmma_bf16(const FragBF& a, const FragBF& b, v8f c) {
  // (neg_a, A, neg_b, B, c_mod, C, reuse_a, reuse_b)
  return __builtin_amdgcn_wmma_f32_16x16x32_bf16(false, a.v, false, b.v,
                                                 (short)0, c, false, false);
}

// ---------------------------------------------------------------------------
// Kernel 0: convert fp32 weights to bf16 scratch copies.
// ---------------------------------------------------------------------------
__global__ void cvt_weights_kernel(const float* __restrict__ qkv_w,
                                   const float* __restrict__ proj_w,
                                   __bf16* __restrict__ qkv_w_bf,
                                   __bf16* __restrict__ proj_w_bf) {
  const int i = blockIdx.x * blockDim.x + threadIdx.x;
  if (i < NQKV * CIN)  qkv_w_bf[i]  = (__bf16)qkv_w[i];
  if (i < COUT * COUT) proj_w_bf[i] = (__bf16)proj_w[i];
}

// ---------------------------------------------------------------------------
// Kernel 1: qkv = x @ qkv_w^T + qkv_b   (M=131072, K=256, N=768)
// Block: 256 threads (8 waves) -> 64-row x 256-col tile (one window per block
// row-wise).  Wave (rg,cg): rows 16, cols 128 -> 8 fp32 C fragments.
// Writes q,k,v bf16, head-major: [bn][h][l][d].
// ---------------------------------------------------------------------------
__global__ __launch_bounds__(256) void qkv_gemm_kernel(
    const float* __restrict__ x, const __bf16* __restrict__ W,
    const float* __restrict__ qkv_b,
    __bf16* __restrict__ Qo, __bf16* __restrict__ Ko, __bf16* __restrict__ Vo) {
  const int lane = threadIdx.x & 31;
  const int wave = threadIdx.x >> 5;
  const int half = lane >> 4;          // 0: lanes 0-15, 1: lanes 16-31
  const int l16  = lane & 15;
  const int bn   = blockIdx.x;                       // window id, 64 rows
  const int m0   = bn * 64 + (wave & 3) * 16;        // wave row base
  const int n0   = blockIdx.y * 256 + (wave >> 2) * 128;

  v8f acc[8];
#pragma unroll
  for (int i = 0; i < 8; ++i) acc[i] = zero8();

  const float* xrow = x + (size_t)(m0 + l16) * CIN;

  for (int k0 = 0; k0 < CIN; k0 += 32) {
    if (k0 + 32 < CIN) __builtin_prefetch(xrow + k0 + 32, 0, 1);
    // A fragment (16x32 bf16): per-lane K chunks [8*half,+8) and [16+8*half,+8)
    FragBF a;
    const float* ap = xrow + k0 + 8 * half;
    const float4 c0 = *(const float4*)(ap);
    const float4 c1 = *(const float4*)(ap + 4);
    const float4 c2 = *(const float4*)(ap + 16);
    const float4 c3 = *(const float4*)(ap + 20);
    a.h[0]  = (__bf16)c0.x; a.h[1]  = (__bf16)c0.y;
    a.h[2]  = (__bf16)c0.z; a.h[3]  = (__bf16)c0.w;
    a.h[4]  = (__bf16)c1.x; a.h[5]  = (__bf16)c1.y;
    a.h[6]  = (__bf16)c1.z; a.h[7]  = (__bf16)c1.w;
    a.h[8]  = (__bf16)c2.x; a.h[9]  = (__bf16)c2.y;
    a.h[10] = (__bf16)c2.z; a.h[11] = (__bf16)c2.w;
    a.h[12] = (__bf16)c3.x; a.h[13] = (__bf16)c3.y;
    a.h[14] = (__bf16)c3.z; a.h[15] = (__bf16)c3.w;
#pragma unroll
    for (int ct = 0; ct < 8; ++ct) {
      const int n = n0 + ct * 16 + l16;
      // B fragment: element(k,n) = W[n][k]; per-lane K chunks [8*half,+8) and [16+8*half,+8).
      FragBF b;
      const __bf16* wp = W + (size_t)n * CIN + k0 + 8 * half;
      b.q[0] = *(const uint4*)(wp);
      b.q[1] = *(const uint4*)(wp + 16);
      acc[ct] = wmma_bf16(a, b, acc[ct]);
    }
  }

  // Epilogue: +bias, scatter into q/k/v bf16 [bn][h][l][d].
#pragma unroll
  for (int ct = 0; ct < 8; ++ct) {
    const int n     = n0 + ct * 16 + l16;
    const int which = n >> 8;
    const int h     = (n >> 5) & 7;
    const int d     = n & 31;
    const float bn_bias = qkv_b[n];
    __bf16* base = (which == 0) ? Qo : ((which == 1) ? Ko : Vo);
    __bf16* dst  = base + ((size_t)(bn * H + h) * L) * HD + d;
#pragma unroll
    for (int r = 0; r < 8; ++r) {
      const int l = (wave & 3) * 16 + r + 8 * half;
      dst[(size_t)l * HD] = (__bf16)(acc[ct][r] + bn_bias);
    }
  }
}

// ---------------------------------------------------------------------------
// Kernel 2: fused attention per (window, head).
// Block: 128 threads (4 waves); wave w owns queries [16w, 16w+16).
// scores = q.k^T (4 WMMA) -> bias/mask/scale -> softmax in registers
// (shfl_xor over 16-lane half-groups) -> p to LDS (A layout) -> p.v (4 WMMA).
// ---------------------------------------------------------------------------
__global__ __launch_bounds__(128) void attn_kernel(
    const __bf16* __restrict__ Qi, const __bf16* __restrict__ Ki,
    const __bf16* __restrict__ Vi, const unsigned char* __restrict__ mask,
    const float* __restrict__ rel_table, const int* __restrict__ rel_index,
    __bf16* __restrict__ attn_out) {
  __shared__ __bf16 vT[HD][L];       // v transposed: vT[d][j]
  __shared__ __bf16 pP[4][16][L];    // per-wave softmax probabilities

  const int bn   = blockIdx.x;       // window id 0..2047
  const int h    = blockIdx.y;       // head
  const int lane = threadIdx.x & 31;
  const int wave = threadIdx.x >> 5;
  const int half = lane >> 4;
  const int l16  = lane & 15;
  const int i0   = wave * 16;

  const __bf16* qb = Qi + (size_t)(bn * H + h) * L * HD;
  const __bf16* kb = Ki + (size_t)(bn * H + h) * L * HD;
  const __bf16* vb = Vi + (size_t)(bn * H + h) * L * HD;

  // Stage v transposed into LDS (64x32 -> 32x64).
  for (int idx = threadIdx.x; idx < L * HD; idx += 128) {
    const int j = idx >> 5;          // row of v
    const int d = idx & 31;          // col of v
    vT[d][j] = vb[idx];
  }
  __syncthreads();

  // --- scores = q . k^T  (A = q rows, B = k^T; contiguous per-lane loads) ---
  FragBF aq;
  const __bf16* qp = qb + (size_t)(i0 + l16) * HD;
  aq.q[0] = *(const uint4*)(qp + 8 * half);
  aq.q[1] = *(const uint4*)(qp + 16 + 8 * half);

  v8f s[4];
#pragma unroll
  for (int jt = 0; jt < 4; ++jt) {
    FragBF bk;
    const __bf16* kp = kb + (size_t)(jt * 16 + l16) * HD + 8 * half;
    bk.q[0] = *(const uint4*)(kp);
    bk.q[1] = *(const uint4*)(kp + 16);
    s[jt] = wmma_bf16(aq, bk, zero8());
  }

  // --- bias + query-mask + scale, then row softmax over the 64 keys. ---
  const float scale = 0.17677669529663687f;   // 1/sqrt(32)
#pragma unroll
  for (int r = 0; r < 8; ++r) {
    const int i = i0 + r + 8 * half;          // query row
    const bool mq = mask[(size_t)bn * L + i] != 0;
    float sv[4];
#pragma unroll
    for (int jt = 0; jt < 4; ++jt) {
      const int j = jt * 16 + l16;
      const float bias = mq ? -100.0f
                            : rel_table[h * NPOS + rel_index[i * L + j]];
      sv[jt] = s[jt][r] * scale + bias;
    }
    float mx = fmaxf(fmaxf(sv[0], sv[1]), fmaxf(sv[2], sv[3]));
#pragma unroll
    for (int off = 1; off < 16; off <<= 1)
      mx = fmaxf(mx, __shfl_xor(mx, off, 32));
    float sum = 0.0f;
#pragma unroll
    for (int jt = 0; jt < 4; ++jt) { sv[jt] = __expf(sv[jt] - mx); sum += sv[jt]; }
#pragma unroll
    for (int off = 1; off < 16; off <<= 1)
      sum += __shfl_xor(sum, off, 32);
    const float inv = 1.0f / sum;
#pragma unroll
    for (int jt = 0; jt < 4; ++jt)
      pP[wave][r + 8 * half][jt * 16 + l16] = (__bf16)(sv[jt] * inv);
  }
  __syncthreads();

  // --- out = p . v  (K = 64 split into two K=32 WMMAs; B from vT in LDS) ---
  v8f o[2] = {zero8(), zero8()};
#pragma unroll
  for (int kh = 0; kh < 2; ++kh) {
    FragBF ap;
    const __bf16* pp = &pP[wave][l16][kh * 32 + 8 * half];
    ap.q[0] = *(const uint4*)(pp);
    ap.q[1] = *(const uint4*)(pp + 16);
#pragma unroll
    for (int dt = 0; dt < 2; ++dt) {
      FragBF bv;
      const __bf16* vp = &vT[dt * 16 + l16][kh * 32 + 8 * half];
      bv.q[0] = *(const uint4*)(vp);
      bv.q[1] = *(const uint4*)(vp + 16);
      o[dt] = wmma_bf16(ap, bv, o[dt]);
    }
  }

  // attn_out[bn][l][h*32+d] (bf16, layout ready for the projection GEMM)
#pragma unroll
  for (int dt = 0; dt < 2; ++dt) {
    const int d = dt * 16 + l16;
    __bf16* ob = attn_out + (size_t)bn * L * COUT + h * HD + d;
#pragma unroll
    for (int r = 0; r < 8; ++r) {
      const int l = i0 + r + 8 * half;
      ob[(size_t)l * COUT] = (__bf16)(o[dt][r]);
    }
  }
}

// ---------------------------------------------------------------------------
// Kernel 3: out = attn_out @ proj_w^T + proj_b   (M=131072, K=256, N=256)
// Block: 256 threads -> 64x256 tile; wave: 16 rows x 128 cols (8 fragments).
// fp32 output straight to d_out.
// ---------------------------------------------------------------------------
__global__ __launch_bounds__(256) void proj_gemm_kernel(
    const __bf16* __restrict__ A, const __bf16* __restrict__ W,
    const float* __restrict__ proj_b, float* __restrict__ out) {
  const int lane = threadIdx.x & 31;
  const int wave = threadIdx.x >> 5;
  const int half = lane >> 4;
  const int l16  = lane & 15;
  const int m0   = blockIdx.x * 64 + (wave & 3) * 16;
  const int n0   = (wave >> 2) * 128;

  v8f acc[8];
#pragma unroll
  for (int i = 0; i < 8; ++i) acc[i] = zero8();

  const __bf16* arow = A + (size_t)(m0 + l16) * COUT;

  for (int k0 = 0; k0 < COUT; k0 += 32) {
    if (k0 + 32 < COUT) __builtin_prefetch(arow + k0 + 32, 0, 1);
    FragBF a;
    const __bf16* ap = arow + k0 + 8 * half;
    a.q[0] = *(const uint4*)(ap);
    a.q[1] = *(const uint4*)(ap + 16);
#pragma unroll
    for (int ct = 0; ct < 8; ++ct) {
      const int n = n0 + ct * 16 + l16;
      FragBF b;
      const __bf16* wp = W + (size_t)n * COUT + k0 + 8 * half;
      b.q[0] = *(const uint4*)(wp);
      b.q[1] = *(const uint4*)(wp + 16);
      acc[ct] = wmma_bf16(a, b, acc[ct]);
    }
  }

#pragma unroll
  for (int ct = 0; ct < 8; ++ct) {
    const int n = n0 + ct * 16 + l16;
    const float bias = proj_b[n];
#pragma unroll
    for (int r = 0; r < 8; ++r) {
      const int m = m0 + r + 8 * half;
      out[(size_t)m * COUT + n] = acc[ct][r] + bias;
    }
  }
}

// ---------------------------------------------------------------------------
// Host-side launch.
// Inputs (setup_inputs order): x, mask, qkv_w, qkv_b, proj_w, proj_b,
//                              rel_table, rel_index
// ---------------------------------------------------------------------------
extern "C" void kernel_launch(void* const* d_in, const int* in_sizes, int n_in,
                              void* d_out, int out_size, void* d_ws, size_t ws_size,
                              hipStream_t stream) {
  const float*         x         = (const float*)d_in[0];
  const unsigned char* mask      = (const unsigned char*)d_in[1]; // jax bool: 1B
  const float*         qkv_w     = (const float*)d_in[2];
  const float*         qkv_b     = (const float*)d_in[3];
  const float*         proj_w    = (const float*)d_in[4];
  const float*         proj_b    = (const float*)d_in[5];
  const float*         rel_table = (const float*)d_in[6];
  const int*           rel_index = (const int*)d_in[7];
  float*               out       = (float*)d_out;

  // Workspace layout (bf16 elements): Q | K | V | attn_out | qkv_w | proj_w
  const size_t QKV_ELEMS = (size_t)BB * NWIN * H * L * HD;   // 33,554,432
  __bf16* Q  = (__bf16*)d_ws;
  __bf16* K  = Q  + QKV_ELEMS;
  __bf16* V  = K  + QKV_ELEMS;
  __bf16* AO = V  + QKV_ELEMS;                               // [bn][l][256]
  __bf16* Wq = AO + QKV_ELEMS;                               // 768*256 bf16
  __bf16* Wp = Wq + (size_t)NQKV * CIN;                      // 256*256 bf16

  // 0) weight conversion
  cvt_weights_kernel<<<(NQKV * CIN + 255) / 256, 256, 0, stream>>>(
      qkv_w, proj_w, Wq, Wp);

  // 1) QKV projection: grid (M/64, 768/256)
  qkv_gemm_kernel<<<dim3(MROWS / 64, NQKV / 256), 256, 0, stream>>>(
      x, Wq, qkv_b, Q, K, V);

  // 2) fused attention: one block per (window, head)
  attn_kernel<<<dim3(BB * NWIN, H), 128, 0, stream>>>(
      Q, K, V, mask, rel_table, rel_index, AO);

  // 3) output projection
  proj_gemm_kernel<<<dim3(MROWS / 64, 1), 256, 0, stream>>>(
      AO, Wp, proj_b, out);
}
